// SparseCINCochainConv_89163521065164
// MI455X (gfx1250) — matrix operand order repaired and checked by reading, not compiled, not run on hardware
//
#include <hip/hip_runtime.h>
#include <stdint.h>

typedef __attribute__((ext_vector_type(16))) __bf16          v16bf;
typedef __attribute__((ext_vector_type(8)))  __bf16          v8bf;
typedef __attribute__((ext_vector_type(8)))  float           v8f;
typedef __attribute__((ext_vector_type(4)))  float           v4f;
typedef __attribute__((ext_vector_type(16))) unsigned short  v16u;
typedef __attribute__((ext_vector_type(4)))  unsigned int    v4u;

#define DFEAT 128

// ---------------------------------------------------------------------------
// bf16 helpers — let the backend pick native cvt instructions
// ---------------------------------------------------------------------------
__device__ __forceinline__ unsigned short f2bf(float f) {
  __bf16 h = (__bf16)f;
  unsigned short u;
  __builtin_memcpy(&u, &h, 2);
  return u;
}

union BF16x16 {
  v16u  u;
  v16bf bf;
  v4u   q[2];
};

// 16 bf16 = 32 contiguous bytes from packed weights -> B fragment
__device__ __forceinline__ v16bf load_bfrag(const unsigned short* p) {
  BF16x16 t;
  t.q[0] = *(const v4u*)(p);
  t.q[1] = *(const v4u*)(p + 8);
  return t.bf;
}

// Build an A fragment (16 bf16) from two 8-float chunks using vector converts
// (backend selects packed f32->bf16 cvt ops).
__device__ __forceinline__ v16bf cvt_afrag(v8f f0, v8f f1) {
  v8bf b0 = __builtin_convertvector(f0, v8bf);
  v8bf b1 = __builtin_convertvector(f1, v8bf);
  return __builtin_shufflevector(b0, b1, 0, 1, 2, 3, 4, 5, 6, 7,
                                 8, 9, 10, 11, 12, 13, 14, 15);
}

__device__ __forceinline__ v8f cat4(v4f a, v4f b) {
  return __builtin_shufflevector(a, b, 0, 1, 2, 3, 4, 5, 6, 7);
}

// A-fragment from f32 row: chunks [p,p+8) and [p+16,p+24)  (regular-temporal)
__device__ __forceinline__ v16bf afrag_f32(const float* p) {
  const v4f a = *(const v4f*)(p);
  const v4f b = *(const v4f*)(p + 4);
  const v4f c = *(const v4f*)(p + 16);
  const v4f d = *(const v4f*)(p + 20);
  return cvt_afrag(cat4(a, b), cat4(c, d));
}

// Same, but non-temporal: for single-use streams (up_attr, 410MB) so they do
// not evict the L2-resident gather targets (x) or the packed weights.
__device__ __forceinline__ v16bf afrag_f32_nt(const float* p) {
  const v4f a = __builtin_nontemporal_load((const v4f*)(p));
  const v4f b = __builtin_nontemporal_load((const v4f*)(p + 4));
  const v4f c = __builtin_nontemporal_load((const v4f*)(p + 16));
  const v4f d = __builtin_nontemporal_load((const v4f*)(p + 20));
  return cvt_afrag(cat4(a, b), cat4(c, d));
}

// ---------------------------------------------------------------------------
// Weight pack: W[K,Nout] f32 row-major -> bf16 B-fragment layout.
// Tile (kt,nt) covers K in [kt*32,kt*32+32), N in [nt*16,nt*16+16).
// Lane L of a tile holds column nt*16+(L&15), K = kt*32 + (t>>3)*16 + (L>>4)*8 + (t&7),
// t = 0..15 stored contiguously (one 32B load per lane at GEMM time).
// ---------------------------------------------------------------------------
__global__ void pack_weights_kernel(const float* __restrict__ W,
                                    unsigned short* __restrict__ Wpk,
                                    int K, int Nout) {
  int lin = blockIdx.x * blockDim.x + threadIdx.x;
  int total = K * Nout;
  if (lin >= total) return;
  int t    = lin & 15;
  int lane = (lin >> 4) & 31;
  int tile = lin >> 9;
  int NT   = Nout >> 4;
  int nt   = tile % NT;
  int kt   = tile / NT;
  int col  = nt * 16 + (lane & 15);
  int k    = kt * 32 + (t >> 3) * 16 + (lane >> 4) * 8 + (t & 7);
  Wpk[lin] = f2bf(W[(size_t)k * Nout + col]);
}

// ---------------------------------------------------------------------------
// Up-message GEMM + scatter:  relu(cat(x[j], up_attr[e]) @ Wmu + bmu)
// atomically accumulated into agg[up_index_i[e]].  One wave = 16 edges x 128
// features: 8 C tiles x 8 K steps = 64 v_wmma_f32_16x16x32_bf16 per wave.
// ---------------------------------------------------------------------------
__global__ void __launch_bounds__(256)
msg_up_kernel(const float* __restrict__ x, const float* __restrict__ up_attr,
              const int* __restrict__ idx_i, const int* __restrict__ idx_j,
              const unsigned short* __restrict__ Wpk,
              const float* __restrict__ bias,
              float* __restrict__ agg, int n_edges) {
  const int lane  = threadIdx.x & 31;
  const int wid   = (blockIdx.x * blockDim.x + threadIdx.x) >> 5;
  const int ebase = wid * 16;
  if (ebase >= n_edges) return;

  const int rA   = lane & 15;
  const int half = lane >> 4;
  const int e    = ebase + rA;
  const int j    = idx_j[e];
  const float* xrow = x + (size_t)j * DFEAT;
  const float* urow = up_attr + (size_t)e * DFEAT;

  v8f acc[8];
#pragma unroll
  for (int nt = 0; nt < 8; ++nt) acc[nt] = 0.0f;

  const int f0 = half * 8;
#pragma unroll
  for (int kk = 0; kk < 8; ++kk) {
    const int kbase = kk * 32;
    // features [0,128): gathered x[j] (keep L2-resident, RT loads)
    // features [128,256): streamed up_attr[e] (non-temporal)
    const v16bf a = (kbase < 128)
                        ? afrag_f32(xrow + kbase + f0)
                        : afrag_f32_nt(urow + (kbase - 128) + f0);
#pragma unroll
    for (int nt = 0; nt < 8; ++nt) {
      const v16bf b = load_bfrag(Wpk + (((kk * 8 + nt) * 32 + lane) << 4));
      acc[nt] = __builtin_amdgcn_wmma_f32_16x16x32_bf16(
          false, a, false, b, (short)0, acc[nt], false, false);
    }
  }

  const int mofs = half * 8;  // C layout: lanes 16-31 hold rows M+8
  int irow[8];
#pragma unroll
  for (int g = 0; g < 8; ++g) irow[g] = idx_i[ebase + g + mofs];

#pragma unroll
  for (int nt = 0; nt < 8; ++nt) {
    const int col = nt * 16 + (lane & 15);
    const float bv = bias[col];
#pragma unroll
    for (int g = 0; g < 8; ++g) {
      float v = acc[nt][g] + bv;
      v = v > 0.0f ? v : 0.0f;
      atomicAdd(agg + (size_t)irow[g] * DFEAT + col, v);
    }
  }
}

// ---------------------------------------------------------------------------
// Dense GEMM (f32 A, K = KT*32), bias + ReLU, f32 output (boundary messages)
// ---------------------------------------------------------------------------
template <int KT>
__global__ void __launch_bounds__(256)
gemm_f32_relu_kernel(const float* __restrict__ A,
                     const unsigned short* __restrict__ Wpk,
                     const float* __restrict__ bias,
                     float* __restrict__ out, int nrows) {
  const int lane    = threadIdx.x & 31;
  const int wid     = (blockIdx.x * blockDim.x + threadIdx.x) >> 5;
  const int rowbase = wid * 16;
  if (rowbase >= nrows) return;

  const int rA   = lane & 15;
  const int half = lane >> 4;
  const float* arow = A + (size_t)(rowbase + rA) * DFEAT;

  v8f acc[8];
#pragma unroll
  for (int nt = 0; nt < 8; ++nt) acc[nt] = 0.0f;

#pragma unroll
  for (int kk = 0; kk < KT; ++kk) {
    const v16bf a = afrag_f32(arow + kk * 32 + half * 8);
#pragma unroll
    for (int nt = 0; nt < 8; ++nt) {
      const v16bf b = load_bfrag(Wpk + (((kk * 8 + nt) * 32 + lane) << 4));
      acc[nt] = __builtin_amdgcn_wmma_f32_16x16x32_bf16(
          false, a, false, b, (short)0, acc[nt], false, false);
    }
  }

  const int mofs = half * 8;
#pragma unroll
  for (int nt = 0; nt < 8; ++nt) {
    const int col = nt * 16 + (lane & 15);
    const float bv = bias[col];
#pragma unroll
    for (int g = 0; g < 8; ++g) {
      float v = acc[nt][g] + bv;
      v = v > 0.0f ? v : 0.0f;
      out[(size_t)(rowbase + g + mofs) * DFEAT + col] = v;
    }
  }
}

// ---------------------------------------------------------------------------
// Boundary incidence scatter: agg_b[i] += msg_b[j]  (one thread per element)
// ---------------------------------------------------------------------------
__global__ void __launch_bounds__(256)
boundary_scatter_kernel(const float* __restrict__ msgB,
                        const int* __restrict__ bidx_i,
                        const int* __restrict__ bidx_j,
                        float* __restrict__ agg, int n_inc) {
  const unsigned tid = blockIdx.x * blockDim.x + threadIdx.x;
  const int e = tid >> 7;
  const int f = tid & 127;
  if (e >= n_inc) return;
  const int i = bidx_i[e];
  const int j = bidx_j[e];
  atomicAdd(agg + (size_t)i * DFEAT + f, msgB[(size_t)j * DFEAT + f]);
}

// ---------------------------------------------------------------------------
// GIN update GEMM: relu((agg + (1+eps)*x) @ W + b), A fused on load, bf16 out
// (feeds the combine GEMM directly as bf16 A).  K = 128.
// ---------------------------------------------------------------------------
__global__ void __launch_bounds__(256)
update_kernel(const float* __restrict__ agg, const float* __restrict__ x,
              const float* __restrict__ eps1,
              const unsigned short* __restrict__ Wpk,
              const float* __restrict__ bias,
              unsigned short* __restrict__ out_bf, int nrows) {
  const int lane    = threadIdx.x & 31;
  const int wid     = (blockIdx.x * blockDim.x + threadIdx.x) >> 5;
  const int rowbase = wid * 16;
  if (rowbase >= nrows) return;

  const float s  = 1.0f + eps1[0];
  const int rA   = lane & 15;
  const int half = lane >> 4;
  const float* arow = agg + (size_t)(rowbase + rA) * DFEAT;
  const float* xrow = x + (size_t)(rowbase + rA) * DFEAT;

  v8f acc[8];
#pragma unroll
  for (int nt = 0; nt < 8; ++nt) acc[nt] = 0.0f;

#pragma unroll
  for (int kk = 0; kk < 4; ++kk) {
    const int f0 = kk * 32 + half * 8;
    v4f h[4];
#pragma unroll
    for (int c = 0; c < 4; ++c) {
      const int fo = f0 + (c >> 1) * 16 + (c & 1) * 4;
      const v4f av = *(const v4f*)(arow + fo);
      const v4f xv = *(const v4f*)(xrow + fo);
      h[c] = av + xv * s;   // agg + (1+eps)*x, elementwise
    }
    const v16bf a = cvt_afrag(cat4(h[0], h[1]), cat4(h[2], h[3]));
#pragma unroll
    for (int nt = 0; nt < 8; ++nt) {
      const v16bf b = load_bfrag(Wpk + (((kk * 8 + nt) * 32 + lane) << 4));
      acc[nt] = __builtin_amdgcn_wmma_f32_16x16x32_bf16(
          false, a, false, b, (short)0, acc[nt], false, false);
    }
  }

  const int mofs = half * 8;
#pragma unroll
  for (int nt = 0; nt < 8; ++nt) {
    const int col = nt * 16 + (lane & 15);
    const float bv = bias[col];
#pragma unroll
    for (int g = 0; g < 8; ++g) {
      float v = acc[nt][g] + bv;
      v = v > 0.0f ? v : 0.0f;
      out_bf[(size_t)(rowbase + g + mofs) * DFEAT + col] = f2bf(v);
    }
  }
}

// ---------------------------------------------------------------------------
// Combine GEMM: relu(cat(out_up, out_b) @ Wc + bc).  A already bf16, K = 256.
// ---------------------------------------------------------------------------
__global__ void __launch_bounds__(256)
combine_kernel(const unsigned short* __restrict__ up_bf,
               const unsigned short* __restrict__ b_bf,
               const unsigned short* __restrict__ Wpk,
               const float* __restrict__ bias,
               float* __restrict__ out, int nrows) {
  const int lane    = threadIdx.x & 31;
  const int wid     = (blockIdx.x * blockDim.x + threadIdx.x) >> 5;
  const int rowbase = wid * 16;
  if (rowbase >= nrows) return;

  const int rA   = lane & 15;
  const int half = lane >> 4;
  const unsigned short* uprow = up_bf + (size_t)(rowbase + rA) * DFEAT;
  const unsigned short* brow  = b_bf + (size_t)(rowbase + rA) * DFEAT;

  v8f acc[8];
#pragma unroll
  for (int nt = 0; nt < 8; ++nt) acc[nt] = 0.0f;

#pragma unroll
  for (int kk = 0; kk < 8; ++kk) {
    const int kbase = kk * 32;
    const unsigned short* src = (kbase < 128) ? uprow : (brow - 128);
    const int f0 = kbase + half * 8;
    BF16x16 t;
    t.q[0] = *(const v4u*)(src + f0);        // 8 bf16: K f0..f0+7
    t.q[1] = *(const v4u*)(src + f0 + 16);   // 8 bf16: K f0+16..f0+23
    const v16bf a = t.bf;
#pragma unroll
    for (int nt = 0; nt < 8; ++nt) {
      const v16bf b = load_bfrag(Wpk + (((kk * 8 + nt) * 32 + lane) << 4));
      acc[nt] = __builtin_amdgcn_wmma_f32_16x16x32_bf16(
          false, a, false, b, (short)0, acc[nt], false, false);
    }
  }

  const int mofs = half * 8;
#pragma unroll
  for (int nt = 0; nt < 8; ++nt) {
    const int col = nt * 16 + (lane & 15);
    const float bv = bias[col];
#pragma unroll
    for (int g = 0; g < 8; ++g) {
      float v = acc[nt][g] + bv;
      v = v > 0.0f ? v : 0.0f;
      out[(size_t)(rowbase + g + mofs) * DFEAT + col] = v;
    }
  }
}

// ---------------------------------------------------------------------------
// Launch
// ---------------------------------------------------------------------------
extern "C" void kernel_launch(void* const* d_in, const int* in_sizes, int n_in,
                              void* d_out, int out_size, void* d_ws, size_t ws_size,
                              hipStream_t stream) {
  const float* x             = (const float*)d_in[0];
  const float* up_attr       = (const float*)d_in[1];
  const float* boundary_attr = (const float*)d_in[2];
  const int*   up_i          = (const int*)d_in[3];
  const int*   up_j          = (const int*)d_in[4];
  const int*   b_i           = (const int*)d_in[5];
  const int*   b_j           = (const int*)d_in[6];
  const float* eps1          = (const float*)d_in[7];
  const float* W_mu = (const float*)d_in[8];
  const float* b_mu = (const float*)d_in[9];
  const float* W_mb = (const float*)d_in[10];
  const float* b_mb = (const float*)d_in[11];
  const float* W_uu = (const float*)d_in[12];
  const float* b_uu = (const float*)d_in[13];
  const float* W_ub = (const float*)d_in[14];
  const float* b_ub = (const float*)d_in[15];
  const float* W_c  = (const float*)d_in[16];
  const float* b_c  = (const float*)d_in[17];

  const int n    = in_sizes[0] / DFEAT;   // 100000
  const int e_up = in_sizes[3];           // 800000
  const int nb   = in_sizes[2] / DFEAT;   // 50000
  const int e_b  = in_sizes[5];           // 400000

  char* ws = (char*)d_ws;
  size_t off = 0;
  auto take = [&](size_t bytes) -> char* {
    char* p = ws + off;
    off = (off + bytes + 255) & ~(size_t)255;
    return p;
  };
  float*          aggUp   = (float*)take((size_t)n * DFEAT * 4);
  float*          aggB    = (float*)take((size_t)n * DFEAT * 4);
  float*          msgB    = (float*)take((size_t)nb * DFEAT * 4);
  unsigned short* outUpBf = (unsigned short*)take((size_t)n * DFEAT * 2);
  unsigned short* outBBf  = (unsigned short*)take((size_t)n * DFEAT * 2);
  unsigned short* WmuPk   = (unsigned short*)take(256 * 128 * 2);
  unsigned short* WmbPk   = (unsigned short*)take(128 * 128 * 2);
  unsigned short* WuuPk   = (unsigned short*)take(128 * 128 * 2);
  unsigned short* WubPk   = (unsigned short*)take(128 * 128 * 2);
  unsigned short* WcPk    = (unsigned short*)take(256 * 128 * 2);

  // 1) pack weights to bf16 WMMA B-fragment layout
  pack_weights_kernel<<<(256 * 128 + 255) / 256, 256, 0, stream>>>(W_mu, WmuPk, 256, 128);
  pack_weights_kernel<<<(128 * 128 + 255) / 256, 256, 0, stream>>>(W_mb, WmbPk, 128, 128);
  pack_weights_kernel<<<(128 * 128 + 255) / 256, 256, 0, stream>>>(W_uu, WuuPk, 128, 128);
  pack_weights_kernel<<<(128 * 128 + 255) / 256, 256, 0, stream>>>(W_ub, WubPk, 128, 128);
  pack_weights_kernel<<<(256 * 128 + 255) / 256, 256, 0, stream>>>(W_c,  WcPk,  256, 128);

  // 2) zero accumulators (graph-capture-legal async memset)
  (void)hipMemsetAsync(aggUp, 0, (size_t)n * DFEAT * 4, stream);
  (void)hipMemsetAsync(aggB,  0, (size_t)n * DFEAT * 4, stream);

  // 3) boundary message GEMM: msgB = relu(boundary_attr @ Wmb + bmb)
  {
    int waves  = (nb + 15) / 16;
    int blocks = (waves + 7) / 8;
    gemm_f32_relu_kernel<4><<<blocks, 256, 0, stream>>>(boundary_attr, WmbPk, b_mb, msgB, nb);
  }

  // 4) up-message GEMM + fused scatter (dominant: 52 GFLOP on WMMA pipe)
  {
    int waves  = (e_up + 15) / 16;
    int blocks = (waves + 7) / 8;
    msg_up_kernel<<<blocks, 256, 0, stream>>>(x, up_attr, up_i, up_j, WmuPk, b_mu, aggUp, e_up);
  }

  // 5) boundary incidence scatter
  {
    long total = (long)e_b * DFEAT;
    int blocks = (int)((total + 255) / 256);
    boundary_scatter_kernel<<<blocks, 256, 0, stream>>>(msgB, b_i, b_j, aggB, e_b);
  }

  // 6) GIN updates (fused agg + (1+eps)x on A load, bf16 out) + combine
  {
    int waves  = (n + 15) / 16;
    int blocks = (waves + 7) / 8;
    update_kernel<<<blocks, 256, 0, stream>>>(aggUp, x, eps1, WuuPk, b_uu, outUpBf, n);
    update_kernel<<<blocks, 256, 0, stream>>>(aggB,  x, eps1, WubPk, b_ub, outBBf, n);
    combine_kernel<<<blocks, 256, 0, stream>>>(outUpBf, outBBf, WcPk, b_c, (float*)d_out, n);
  }
}
